// SSDeformAttnMyVersionV2_59133109731752
// MI455X (gfx1250) — matrix operand order repaired and checked by reading, not compiled, not run on hardware
//
#include <hip/hip_runtime.h>
#include <hip/hip_bf16.h>

typedef __attribute__((ext_vector_type(16))) _Float16     v16h;
typedef __attribute__((ext_vector_type(8)))  float        v8f;
typedef __attribute__((ext_vector_type(4)))  unsigned int v4u;

#define HH 80
#define WW 80
#define HWSZ 6400
#define NBATCH 4
#define PAD 40   // LDS channel-stride in halves: 80B -> 16B-aligned frags, conflict-free

// ---------------- weight packing: f32 -> f16 in WMMA-friendly [chunk][co][32] layout ----
// 3x3 conv: K ordering k = tap*Cin + ci ; chunk = tap*(Cin/32) + s ; element kl = ci%32
__global__ void pack3x3_kernel(const float* __restrict__ w, _Float16* __restrict__ wpk,
                               int Cin, int Cout) {
  int idx = blockIdx.x * 256 + threadIdx.x;
  int total = Cout * Cin * 9;
  if (idx >= total) return;
  int kl    = idx & 31;
  int co    = (idx >> 5) % Cout;
  int chunk = idx / (32 * Cout);
  int nchunk = Cin >> 5;
  int s = chunk % nchunk;
  int t = chunk / nchunk;           // tap = dy*3+dx
  int ci = s * 32 + kl;
  wpk[idx] = (_Float16)w[(co * Cin + ci) * 9 + t];
}

__global__ void pack1x1_kernel(const float* __restrict__ w, _Float16* __restrict__ wpk,
                               int Cin, int Cout) {
  int idx = blockIdx.x * 256 + threadIdx.x;
  int total = Cout * Cin;
  if (idx >= total) return;
  int kl = idx & 31;
  int co = (idx >> 5) % Cout;
  int s  = idx / (32 * Cout);
  wpk[idx] = (_Float16)w[co * Cin + s * 32 + kl];
}

__device__ __forceinline__ unsigned pack2h(float a, float b) {
  union { unsigned u; _Float16 h[2]; } pk;
  pk.h[0] = (_Float16)a;
  pk.h[1] = (_Float16)b;
  return pk.u;
}

// ---------------- 3x3 conv, implicit GEMM via v_wmma_f32_16x16x32_f16 -------------------
// block = 256 thr (8 waves) handles one (n,y) row: 80 pixels x 128 cout.
// wave w -> cout tile w*16 ; loops 5 pixel tiles of 16.
// mode: 0=none, 1=leaky(0.1), 2=10*tanh
__global__ __launch_bounds__(256)
void conv3x3_wmma_kernel(const float* __restrict__ in, const _Float16* __restrict__ wpk,
                         const float* __restrict__ bias, float* __restrict__ out,
                         int Cin, int mode) {
  __shared__ _Float16 patch[3 * 82 * PAD];  // [row][col][ci], ci stride padded to 40 halves
  const int tid  = threadIdx.x;
  const int lane = tid & 31;
  const int wave = tid >> 5;
  const int n = blockIdx.x / HH;
  const int y = blockIdx.x % HH;
  const int nchunk = Cin >> 5;

  const int m    = lane & 15;               // A row / C column index
  const int koff = (lane < 16) ? 0 : 16;    // A-frag K offset (halves)
  const int bko  = (lane < 16) ? 0 : 16;    // B-frag K offset (halves)
  const int co   = wave * 16 + m;

  v8f acc[5];
  const v8f vzero = {0.f,0.f,0.f,0.f,0.f,0.f,0.f,0.f};
#pragma unroll
  for (int i = 0; i < 5; ++i) acc[i] = vzero;

  for (int s = 0; s < nchunk; ++s) {
    __syncthreads();
    // stage 3 input rows x 82 cols x 32 channels (zero-padded borders), 2 ci per dword
    for (int idx = tid; idx < 3 * 82 * 16; idx += 256) {
      int col = idx % 82;
      int row = (idx / 82) % 3;
      int cq  = idx / (82 * 3);             // channel pair 0..15
      int gy = y + row - 1;
      int gx = col - 1;
      float v0 = 0.f, v1 = 0.f;
      if (gy >= 0 && gy < HH && gx >= 0 && gx < WW) {
        const float* ip = in + ((n * Cin + s * 32 + 2 * cq) * HH + gy) * WW + gx;
        v0 = ip[0];
        v1 = ip[HH * WW];
      }
      *(unsigned*)(&patch[(row * 82 + col) * PAD + 2 * cq]) = pack2h(v0, v1);
    }
    __syncthreads();
#pragma unroll
    for (int t = 0; t < 9; ++t) {
      const int dy = t / 3, dx = t % 3;
      const int chunk = t * nchunk + s;
      v16h bfrag = *(const v16h*)(wpk + (size_t)(chunk * 128 + co) * 32 + bko);
#pragma unroll
      for (int pt = 0; pt < 5; ++pt) {
        const int x = pt * 16 + m;
        const int base = (dy * 82 + (x + dx)) * PAD + koff;  // 16B-aligned
        union { v16h v; v4u q[2]; } af;
        af.q[0] = *(const v4u*)(&patch[base]);        // ds_load_b128
        af.q[1] = *(const v4u*)(&patch[base + 8]);    // ds_load_b128
        acc[pt] = __builtin_amdgcn_wmma_f32_16x16x32_f16(
            false, af.v, false, bfrag, (short)0, acc[pt], false, false);
      }
    }
  }
  const float bb = bias[co];
  const int mrow = (lane >> 4) << 3;
#pragma unroll
  for (int pt = 0; pt < 5; ++pt) {
#pragma unroll
    for (int i = 0; i < 8; ++i) {
      const int x = pt * 16 + mrow + i;
      float r = acc[pt][i] + bb;
      if (mode == 1)      r = (r >= 0.f) ? r : 0.1f * r;
      else if (mode == 2) r = 10.f * tanhf(r);
      out[((n * 128 + co) * HH + y) * WW + x] = r;
    }
  }
}

// ---------------- 1x1 conv via WMMA: block = 64 pixels x 128 cout -----------------------
__global__ __launch_bounds__(256)
void conv1x1_wmma_kernel(const float* __restrict__ in, const _Float16* __restrict__ wpk,
                         const float* __restrict__ bias, float* __restrict__ out, int Cin) {
  __shared__ _Float16 ap[64 * PAD];
  const int tid  = threadIdx.x;
  const int lane = tid & 31;
  const int wave = tid >> 5;
  const int p0 = blockIdx.x * 64;           // global pixel over N*H*W (6400 % 64 == 0)
  const int n  = p0 / HWSZ;
  const int pb = p0 % HWSZ;
  const int nchunk = Cin >> 5;

  const int m    = lane & 15;
  const int koff = (lane < 16) ? 0 : 16;
  const int bko  = (lane < 16) ? 0 : 16;
  const int co   = wave * 16 + m;

  v8f acc[4];
  const v8f vzero = {0.f,0.f,0.f,0.f,0.f,0.f,0.f,0.f};
#pragma unroll
  for (int i = 0; i < 4; ++i) acc[i] = vzero;

  for (int s = 0; s < nchunk; ++s) {
    __syncthreads();
    for (int idx = tid; idx < 64 * 16; idx += 256) {
      int px = idx & 63;
      int cq = idx >> 6;                    // channel pair 0..15
      const float* ip = in + (size_t)(n * Cin + s * 32 + 2 * cq) * HWSZ + pb + px;
      *(unsigned*)(&ap[px * PAD + 2 * cq]) = pack2h(ip[0], ip[HWSZ]);
    }
    __syncthreads();
    v16h bfrag = *(const v16h*)(wpk + (size_t)(s * 128 + co) * 32 + bko);
#pragma unroll
    for (int pt = 0; pt < 4; ++pt) {
      const int base = (pt * 16 + m) * PAD + koff;
      union { v16h v; v4u q[2]; } af;
      af.q[0] = *(const v4u*)(&ap[base]);
      af.q[1] = *(const v4u*)(&ap[base + 8]);
      acc[pt] = __builtin_amdgcn_wmma_f32_16x16x32_f16(
          false, af.v, false, bfrag, (short)0, acc[pt], false, false);
    }
  }
  const float bb = bias[co];
  const int mrow = (lane >> 4) << 3;
#pragma unroll
  for (int pt = 0; pt < 4; ++pt) {
#pragma unroll
    for (int i = 0; i < 8; ++i) {
      out[(size_t)(n * 128 + co) * HWSZ + pb + pt * 16 + mrow + i] = acc[pt][i] + bb;
    }
  }
}

// ---------------- deformable sampling + softmax attention -------------------------------
// one thread per (n, head, pixel). Reshape quirk: for (h,pt), off_x = flatblock[2q],
// off_y = flatblock[2q+1] where flatblock = channels [2(h*8+pt), 2(h*8+pt)+1] flattened.
__global__ __launch_bounds__(256)
void deform_attn_kernel(const float* __restrict__ v, const float* __restrict__ ref,
                        const float* __restrict__ off, float* __restrict__ out) {
  int gid = blockIdx.x * 256 + threadIdx.x;
  if (gid >= 32 * HWSZ) return;
  const int p  = gid % HWSZ;
  const int nh = gid / HWSZ;
  const int n = nh >> 3, h = nh & 7;

  const float* vh = v   + (size_t)(n * 128 + h * 16) * HWSZ;
  const float* ob = off + (size_t)n * 128 * HWSZ;

  float vc[16];
#pragma unroll
  for (int d = 0; d < 16; ++d) vc[d] = vh[d * HWSZ + p];

  const float rx = ref[(size_t)(n * HWSZ + p) * 2 + 0];
  const float ry = ref[(size_t)(n * HWSZ + p) * 2 + 1];

  float sxa[8], sya[8], sc[8];
#pragma unroll
  for (int pt = 0; pt < 8; ++pt) {
    const size_t blk = (size_t)((h * 8 + pt) * 2) * HWSZ;
    const float ox = ob[blk + 2 * p + 0];
    const float oy = ob[blk + 2 * p + 1];
    const float sx = (rx + ox * (1.0f / WW)) * WW - 0.5f;  // = loc_x*W - 0.5
    const float sy = (ry + oy * (1.0f / HH)) * HH - 0.5f;
    sxa[pt] = sx; sya[pt] = sy;
    const float x0f = floorf(sx), y0f = floorf(sy);
    const int x0 = (int)x0f, y0 = (int)y0f;
    const float wx1 = sx - x0f, wy1 = sy - y0f;
    const float wx0 = 1.f - wx1, wy0 = 1.f - wy1;
    const int   xs[4] = {x0, x0 + 1, x0,     x0 + 1};
    const int   ys[4] = {y0, y0,     y0 + 1, y0 + 1};
    const float wt[4] = {wx0 * wy0, wx1 * wy0, wx0 * wy1, wx1 * wy1};
    float s = 0.f;
#pragma unroll
    for (int tap = 0; tap < 4; ++tap) {
      const int xi = xs[tap], yi = ys[tap];
      if (xi >= 0 && xi < WW && yi >= 0 && yi < HH) {
        const float* pp = vh + yi * WW + xi;
        float dd = 0.f;
#pragma unroll
        for (int d = 0; d < 16; ++d) dd += vc[d] * pp[d * HWSZ];
        s += wt[tap] * dd;
      }
    }
    sc[pt] = s;
  }
  // softmax over 8 points
  float mx = sc[0];
#pragma unroll
  for (int pt = 1; pt < 8; ++pt) mx = fmaxf(mx, sc[pt]);
  float sum = 0.f;
#pragma unroll
  for (int pt = 0; pt < 8; ++pt) { sc[pt] = expf(sc[pt] - mx); sum += sc[pt]; }
  const float inv = 1.f / sum;

  float o16[16];
#pragma unroll
  for (int d = 0; d < 16; ++d) o16[d] = 0.f;
#pragma unroll
  for (int pt = 0; pt < 8; ++pt) {
    const float aw = sc[pt] * inv;
    const float sx = sxa[pt], sy = sya[pt];
    const float x0f = floorf(sx), y0f = floorf(sy);
    const int x0 = (int)x0f, y0 = (int)y0f;
    const float wx1 = sx - x0f, wy1 = sy - y0f;
    const float wx0 = 1.f - wx1, wy0 = 1.f - wy1;
    const int   xs[4] = {x0, x0 + 1, x0,     x0 + 1};
    const int   ys[4] = {y0, y0,     y0 + 1, y0 + 1};
    const float wt[4] = {aw * wx0 * wy0, aw * wx1 * wy0, aw * wx0 * wy1, aw * wx1 * wy1};
#pragma unroll
    for (int tap = 0; tap < 4; ++tap) {
      const int xi = xs[tap], yi = ys[tap];
      if (xi >= 0 && xi < WW && yi >= 0 && yi < HH) {
        const float* pp = vh + yi * WW + xi;
#pragma unroll
        for (int d = 0; d < 16; ++d) o16[d] += wt[tap] * pp[d * HWSZ];
      }
    }
  }
#pragma unroll
  for (int d = 0; d < 16; ++d)
    out[(size_t)(n * 128 + h * 16 + d) * HWSZ + p] = o16[d];
}

// ---------------- launch ----------------------------------------------------------------
extern "C" void kernel_launch(void* const* d_in, const int* in_sizes, int n_in,
                              void* d_out, int out_size, void* d_ws, size_t ws_size,
                              hipStream_t stream) {
  const float* query = (const float*)d_in[0];
  const float* value = (const float*)d_in[1];
  const float* refp  = (const float*)d_in[2];
  const float* co_w1 = (const float*)d_in[3];
  const float* co_b1 = (const float*)d_in[4];
  const float* co_w2 = (const float*)d_in[5];
  const float* co_b2 = (const float*)d_in[6];
  const float* co_w3 = (const float*)d_in[7];
  const float* co_b3 = (const float*)d_in[8];
  const float* co_w4 = (const float*)d_in[9];
  const float* co_b4 = (const float*)d_in[10];
  const float* val_w = (const float*)d_in[11];
  const float* val_b = (const float*)d_in[12];
  const float* out_w = (const float*)d_in[13];
  const float* out_b = (const float*)d_in[14];

  char* ws = (char*)d_ws;
  const size_t ACT = (size_t)NBATCH * 128 * HWSZ * sizeof(float);  // 13107200
  float*    v_buf = (float*)(ws);
  float*    bufA  = (float*)(ws + ACT);
  float*    bufB  = (float*)(ws + 2 * ACT);
  _Float16* wpk1  = (_Float16*)(ws + 3 * ACT);                       // 2304*128 f16
  _Float16* wpk2  = (_Float16*)(ws + 3 * ACT +  589824);
  _Float16* wpk3  = (_Float16*)(ws + 3 * ACT +  884736);
  _Float16* wpk4  = (_Float16*)(ws + 3 * ACT + 1179648);
  _Float16* wpkV  = (_Float16*)(ws + 3 * ACT + 1474560);
  _Float16* wpkO  = (_Float16*)(ws + 3 * ACT + 1507328);

  // pack weights (f32 -> f16, WMMA B layout)
  pack3x3_kernel<<<(128 * 256 * 9 + 255) / 256, 256, 0, stream>>>(co_w1, wpk1, 256, 128);
  pack3x3_kernel<<<(128 * 128 * 9 + 255) / 256, 256, 0, stream>>>(co_w2, wpk2, 128, 128);
  pack3x3_kernel<<<(128 * 128 * 9 + 255) / 256, 256, 0, stream>>>(co_w3, wpk3, 128, 128);
  pack3x3_kernel<<<(128 * 128 * 9 + 255) / 256, 256, 0, stream>>>(co_w4, wpk4, 128, 128);
  pack1x1_kernel<<<(128 * 128 + 255) / 256, 256, 0, stream>>>(val_w, wpkV, 128, 128);
  pack1x1_kernel<<<(128 * 128 + 255) / 256, 256, 0, stream>>>(out_w, wpkO, 128, 128);

  // value projection
  conv1x1_wmma_kernel<<<(NBATCH * HWSZ) / 64, 256, 0, stream>>>(value, wpkV, val_b, v_buf, 128);
  // offset network
  conv3x3_wmma_kernel<<<NBATCH * HH, 256, 0, stream>>>(query, wpk1, co_b1, bufA, 256, 1);
  conv3x3_wmma_kernel<<<NBATCH * HH, 256, 0, stream>>>(bufA,  wpk2, co_b2, bufB, 128, 1);
  conv3x3_wmma_kernel<<<NBATCH * HH, 256, 0, stream>>>(bufB,  wpk3, co_b3, bufA, 128, 1);
  conv3x3_wmma_kernel<<<NBATCH * HH, 256, 0, stream>>>(bufA,  wpk4, co_b4, bufB, 128, 2);
  // deformable attention (v_buf, bufB=offsets) -> bufA
  deform_attn_kernel<<<(32 * HWSZ + 255) / 256, 256, 0, stream>>>(v_buf, refp, bufB, bufA);
  // output projection
  conv1x1_wmma_kernel<<<(NBATCH * HWSZ) / 64, 256, 0, stream>>>(bufA, wpkO, out_b, (float*)d_out, 128);
}
